// CrissCrossAttention_Ref_36524401885630
// MI455X (gfx1250) — matrix operand order repaired and checked by reading, not compiled, not run on hardware
//
#include <hip/hip_runtime.h>
#include <hip/hip_bf16.h>

// CrissCrossAttention on gfx1250: all matmuls via v_wmma_f32_16x16x32_bf16
// (bf16 operands, f32 accumulate). Roofline: ~65 GFLOP, ~1 GB HBM traffic
// -> bf16 WMMA keeps matrix math under the 23.3 TB/s memory roofline.

#define BB   8
#define CC   512
#define C8d  64
#define HHd  96
#define WWd  96
#define HWP  (HHd * WWd)   // 9216

typedef __attribute__((ext_vector_type(16))) __bf16 v16bf;
typedef __attribute__((ext_vector_type(8)))  float  v8f;

union Frag {
  v16bf v;
  uint4 u[2];
  unsigned short h[16];
  unsigned d[8];
};

__device__ __forceinline__ unsigned short f2bf(float f) {
  unsigned u = __float_as_uint(f);
  u += 0x7FFFu + ((u >> 16) & 1u);     // round-to-nearest-even
  return (unsigned short)(u >> 16);
}

// pack two f32 -> one dword of two bf16 (lo = a, hi = b)
__device__ __forceinline__ unsigned pk_bf16(float a, float b) {
#if __has_builtin(__builtin_amdgcn_cvt_pk_bf16_f32)
  typedef __attribute__((ext_vector_type(2))) __bf16 v2bf;
  union { v2bf v; unsigned d; } r;
  r.v = __builtin_amdgcn_cvt_pk_bf16_f32(a, b);
  return r.d;
#else
  return (unsigned)f2bf(a) | ((unsigned)f2bf(b) << 16);
#endif
}

__device__ __forceinline__ v8f wmma_bf16(const Frag& a, const Frag& b, v8f c) {
  // D = A(16x32) * B(32x16) + C, f32 accumulate
  return __builtin_amdgcn_wmma_f32_16x16x32_bf16(false, a.v, false, b.v,
                                                 (short)0, c, false, false);
}

// ---------------------------------------------------------------------------
// Projection GEMM: out[o,p] = sum_c W[o,c] * X[c,p] + bias[o]
// MODE 0 (q/k, O=64,  MT=4):  store bf16 to dstA[((b*W+w)*H+h)*64+o] and
//                                          dstB[((b*H+h)*W+w)*64+o]
// MODE 1 (v,  O=512, MT=16): store bf16 natural dstA[(b*C+o)*HW + p]
// Block: 256 thr (8 waves), N=128 pixels/block, K double-buffered via LDS.
// All tile loops have compile-time trip counts -> no scalar branching in the
// WMMA inner loop.
// ---------------------------------------------------------------------------
template <int MT, int MODE>
__global__ void __launch_bounds__(256)
proj_kernel(const float* __restrict__ X, const float* __restrict__ Wm,
            const float* __restrict__ bias,
            unsigned short* __restrict__ dstA, unsigned short* __restrict__ dstB)
{
  __shared__ unsigned Xl[2][128 * 20];  // [buf][p*20 + c2]: 2 bf16 packed, pitch 40 halves

  const int tid  = threadIdx.x;
  const int wave = tid >> 5;
  const int lane = tid & 31;
  const int ln   = lane & 15;
  const int sel  = lane >> 4;

  const int blk = blockIdx.x;
  const int b   = blk / 72;
  const int p0  = (blk % 72) * 128;
  const float* Xb = X + (size_t)b * CC * HWP;

  const int plocal = wave * 16 + ln;
  const int p      = p0 + plocal;
  const int ph     = p / WWd;
  const int pw     = p % WWd;

  auto stage = [&](int ks, int buf) {
    const int kbase = ks * 32;
#pragma unroll
    for (int i = 0; i < 8; ++i) {
      const int idx = tid + (i << 8);
      const int c2  = idx >> 7;       // 0..15
      const int pp  = idx & 127;
      const int c   = kbase + (c2 << 1);
      const float x0 = Xb[(size_t)c * HWP + p0 + pp];
      const float x1 = Xb[(size_t)(c + 1) * HWP + p0 + pp];
      Xl[buf][pp * 20 + c2] = pk_bf16(x0, x1);
    }
  };

  constexpr int MTILES = (MODE == 0) ? 4 : 32;   // O/16
  static_assert(MTILES % MT == 0, "exact passes");

  for (int m0 = 0; m0 < MTILES; m0 += MT) {
    v8f acc[MT];
#pragma unroll
    for (int mi = 0; mi < MT; ++mi) {
      const int ob = (m0 + mi) * 16 + 8 * sel;
      v8f c;
#pragma unroll
      for (int j = 0; j < 8; ++j) c[j] = bias[ob + j];
      acc[mi] = c;
    }

    stage(0, 0);
    __syncthreads();
    for (int ks = 0; ks < 16; ++ks) {
      if (ks + 1 < 16) stage(ks + 1, (ks + 1) & 1);

      const unsigned short* Xh = (const unsigned short*)Xl[ks & 1];
      Frag bfr;
      const int boff = plocal * 40 + 8 * sel;
      bfr.u[0] = *(const uint4*)(Xh + boff);
      bfr.u[1] = *(const uint4*)(Xh + boff + 16);

#pragma unroll
      for (int mi = 0; mi < MT; ++mi) {
        const int o = (m0 + mi) * 16 + ln;
        const float* wrow = Wm + (size_t)o * CC + ks * 32;
        const float4 f0 = *(const float4*)(wrow + 8 * sel);
        const float4 f1 = *(const float4*)(wrow + 8 * sel + 4);
        const float4 f2 = *(const float4*)(wrow + 16 + 8 * sel);
        const float4 f3 = *(const float4*)(wrow + 16 + 8 * sel + 4);
        Frag a;
        a.d[0] = pk_bf16(f0.x, f0.y);
        a.d[1] = pk_bf16(f0.z, f0.w);
        a.d[2] = pk_bf16(f1.x, f1.y);
        a.d[3] = pk_bf16(f1.z, f1.w);
        a.d[4] = pk_bf16(f2.x, f2.y);
        a.d[5] = pk_bf16(f2.z, f2.w);
        a.d[6] = pk_bf16(f3.x, f3.y);
        a.d[7] = pk_bf16(f3.z, f3.w);
        acc[mi] = wmma_bf16(a, bfr, acc[mi]);
      }
      __syncthreads();
    }

#pragma unroll
    for (int mi = 0; mi < MT; ++mi) {
      const int otile = (m0 + mi) * 16;
      if (MODE == 0) {
        union { uint4 u; unsigned d[4]; } pk;
#pragma unroll
        for (int j = 0; j < 4; ++j)
          pk.d[j] = pk_bf16(acc[mi][2 * j], acc[mi][2 * j + 1]);
        const size_t cb = (size_t)(otile + 8 * sel);
        const size_t aA = ((size_t)(b * WWd + pw) * HHd + ph) * C8d + cb;
        const size_t aB = ((size_t)(b * HHd + ph) * WWd + pw) * C8d + cb;
        *(uint4*)(dstA + aA) = pk.u;
        *(uint4*)(dstB + aB) = pk.u;
      } else {
#pragma unroll
        for (int r = 0; r < 8; ++r) {
          const int o = otile + r + 8 * sel;
          dstA[(size_t)(b * CC + o) * HWP + p] = f2bf(acc[mi][r]);
        }
      }
    }
  }
}

// ---------------------------------------------------------------------------
// bf16 (b,c,h,w) -> (b,c,w,h) transpose via LDS, one 96x96 plane per block
// ---------------------------------------------------------------------------
__global__ void __launch_bounds__(256)
transpose_hw_kernel(const unsigned short* __restrict__ src,
                    unsigned short* __restrict__ dst)
{
  __shared__ unsigned short t[HHd * 97];
  const size_t plane = (size_t)blockIdx.x * HWP;
  for (int idx = threadIdx.x; idx < HWP; idx += 256) {
    const int h = idx / WWd, w = idx % WWd;
    t[w * 97 + h] = src[plane + idx];
  }
  __syncthreads();
  for (int idx = threadIdx.x; idx < HWP; idx += 256) {
    const int w = idx / HHd, h = idx % HHd;
    dst[plane + idx] = t[w * 97 + h];
  }
}

// ---------------------------------------------------------------------------
// Energy: per slice s, E[m,n] = sum_c q[s,m,c] * k[s,n,c]  (96x96x64 GEMM)
// variant 0: s=w (column attention, diag masked -inf) ; variant 1: s=h (row)
// Block: 192 thr (6 waves), wave = m-tile, 6 n-tiles x 2 k-steps each.
// ---------------------------------------------------------------------------
__global__ void __launch_bounds__(192)
energy_kernel(const unsigned short* __restrict__ qA, const unsigned short* __restrict__ kA,
              const unsigned short* __restrict__ qB, const unsigned short* __restrict__ kB,
              float* __restrict__ EH, float* __restrict__ EW)
{
  const int id      = blockIdx.x;
  const int variant = (id >= BB * WWd) ? 1 : 0;
  const int rem     = variant ? id - BB * WWd : id;
  const int b       = rem / WWd;
  const int s       = rem % WWd;

  const unsigned short* qb = variant ? qB : qA;
  const unsigned short* kb = variant ? kB : kA;
  float* E = variant ? EW : EH;

  const int lane = threadIdx.x & 31;
  const int ln   = lane & 15;
  const int sel  = lane >> 4;
  const int mt   = threadIdx.x >> 5;       // 0..5

  const size_t sbase = (size_t)(b * 96 + s) * 96;

  const v8f z = {0.f, 0.f, 0.f, 0.f, 0.f, 0.f, 0.f, 0.f};
  v8f acc[6];
#pragma unroll
  for (int nt = 0; nt < 6; ++nt) acc[nt] = z;

#pragma unroll
  for (int kk = 0; kk < 64; kk += 32) {
    Frag a;
    const unsigned short* ap = qb + (sbase + mt * 16 + ln) * C8d + kk + 8 * sel;
    a.u[0] = *(const uint4*)ap;
    a.u[1] = *(const uint4*)(ap + 16);
#pragma unroll
    for (int nt = 0; nt < 6; ++nt) {
      Frag bf;
      const unsigned short* bp = kb + (sbase + nt * 16 + ln) * C8d + kk + 8 * sel;
      bf.u[0] = *(const uint4*)bp;
      bf.u[1] = *(const uint4*)(bp + 16);
      acc[nt] = wmma_bf16(a, bf, acc[nt]);
    }
  }

#pragma unroll
  for (int nt = 0; nt < 6; ++nt) {
#pragma unroll
    for (int r = 0; r < 8; ++r) {
      const int m = mt * 16 + r + 8 * sel;
      const int n = nt * 16 + ln;
      float v = acc[nt][r];
      if (!variant && m == n) v = -__builtin_inff();
      E[(sbase + m) * 96 + n] = v;
    }
  }
}

// ---------------------------------------------------------------------------
// Joint softmax over 192 logits/pixel (96 column + 96 row); one wave/pixel.
// ---------------------------------------------------------------------------
__global__ void __launch_bounds__(256)
softmax_kernel(const float* __restrict__ EH, const float* __restrict__ EW,
               unsigned short* __restrict__ aH, unsigned short* __restrict__ aW)
{
  const int wave = threadIdx.x >> 5;
  const int lane = threadIdx.x & 31;
  const int pix  = blockIdx.x * 8 + wave;
  const int b    = pix / HWP;
  const int r    = pix % HWP;
  const int h    = r / WWd;
  const int w    = r % WWd;

  const size_t baseH = ((size_t)(b * WWd + w) * HHd + h) * 96;
  const size_t baseW = ((size_t)(b * HHd + h) * WWd + w) * 96;

  float v[6];
#pragma unroll
  for (int i = 0; i < 3; ++i) v[i]     = EH[baseH + lane + 32 * i];
#pragma unroll
  for (int i = 0; i < 3; ++i) v[3 + i] = EW[baseW + lane + 32 * i];

  float m = v[0];
#pragma unroll
  for (int i = 1; i < 6; ++i) m = fmaxf(m, v[i]);
#pragma unroll
  for (int off = 16; off >= 1; off >>= 1) m = fmaxf(m, __shfl_xor(m, off, 32));

  float e[6], s = 0.f;
#pragma unroll
  for (int i = 0; i < 6; ++i) { e[i] = __expf(v[i] - m); s += e[i]; }
#pragma unroll
  for (int off = 16; off >= 1; off >>= 1) s += __shfl_xor(s, off, 32);
  const float inv = 1.0f / s;

#pragma unroll
  for (int i = 0; i < 3; ++i) aH[baseH + lane + 32 * i] = f2bf(e[i] * inv);
#pragma unroll
  for (int i = 0; i < 3; ++i) aW[baseW + lane + 32 * i] = f2bf(e[3 + i] * inv);
}

// ---------------------------------------------------------------------------
// out_H[b,c,h,w] = sum_g v[b,c,g,w] * attH[b,h,w,g]   (512x96x96 per (b,w))
// A from v_t[b][c][w][g] (g contiguous), B from aH[b][w][h][g]. Writes d_out.
// ---------------------------------------------------------------------------
__global__ void __launch_bounds__(256)
agg_h_kernel(const unsigned short* __restrict__ vt, const unsigned short* __restrict__ aH,
             float* __restrict__ out)
{
  const int b    = blockIdx.x / WWd;
  const int w    = blockIdx.x % WWd;
  const int wave = threadIdx.x >> 5;
  const int lane = threadIdx.x & 31;
  const int ln   = lane & 15;
  const int sel  = lane >> 4;

  const size_t attbase = (size_t)(b * WWd + w) * HHd * 96;
  const v8f z = {0.f, 0.f, 0.f, 0.f, 0.f, 0.f, 0.f, 0.f};

#pragma unroll
  for (int mi = 0; mi < 4; ++mi) {
    const int mt = wave * 4 + mi;            // c-tile 0..31
    v8f acc[6];
#pragma unroll
    for (int nt = 0; nt < 6; ++nt) acc[nt] = z;

#pragma unroll
    for (int kk = 0; kk < 96; kk += 32) {
      Frag a;
      const int c = mt * 16 + ln;
      const unsigned short* ap =
          vt + ((size_t)(b * CC + c) * 96 + w) * 96 + kk + 8 * sel;
      a.u[0] = *(const uint4*)ap;
      a.u[1] = *(const uint4*)(ap + 16);
#pragma unroll
      for (int nt = 0; nt < 6; ++nt) {
        Frag bf;
        const unsigned short* bp =
            aH + attbase + (size_t)(nt * 16 + ln) * 96 + kk + 8 * sel;
        bf.u[0] = *(const uint4*)bp;
        bf.u[1] = *(const uint4*)(bp + 16);
        acc[nt] = wmma_bf16(a, bf, acc[nt]);
      }
    }

#pragma unroll
    for (int nt = 0; nt < 6; ++nt) {
#pragma unroll
      for (int r = 0; r < 8; ++r) {
        const int c  = mt * 16 + r + 8 * sel;
        const int hh = nt * 16 + ln;
        out[(size_t)(b * CC + c) * HWP + hh * WWd + w] = acc[nt][r];
      }
    }
  }
}

// ---------------------------------------------------------------------------
// out_W + epilogue: d_out = gamma*(out_H + out_W) + f_low  (512x96x96 per (b,h))
// A from bf16 v natural layout (w' contiguous), B from aW[b][h][w][w'].
// ---------------------------------------------------------------------------
__global__ void __launch_bounds__(256)
agg_w_kernel(const unsigned short* __restrict__ vn, const unsigned short* __restrict__ aW,
             const float* __restrict__ f_low, const float* __restrict__ gamma,
             float* __restrict__ out)
{
  const int b    = blockIdx.x / HHd;
  const int h    = blockIdx.x % HHd;
  const int wave = threadIdx.x >> 5;
  const int lane = threadIdx.x & 31;
  const int ln   = lane & 15;
  const int sel  = lane >> 4;

  const float g = gamma[0];
  const size_t attbase = (size_t)(b * HHd + h) * WWd * 96;
  const v8f z = {0.f, 0.f, 0.f, 0.f, 0.f, 0.f, 0.f, 0.f};

#pragma unroll
  for (int mi = 0; mi < 4; ++mi) {
    const int mt = wave * 4 + mi;
    v8f acc[6];
#pragma unroll
    for (int nt = 0; nt < 6; ++nt) acc[nt] = z;

#pragma unroll
    for (int kk = 0; kk < 96; kk += 32) {
      Frag a;
      const int c = mt * 16 + ln;
      const unsigned short* ap =
          vn + (size_t)(b * CC + c) * HWP + h * WWd + kk + 8 * sel;
      a.u[0] = *(const uint4*)ap;
      a.u[1] = *(const uint4*)(ap + 16);
#pragma unroll
      for (int nt = 0; nt < 6; ++nt) {
        Frag bf;
        const unsigned short* bp =
            aW + attbase + (size_t)(nt * 16 + ln) * 96 + kk + 8 * sel;
        bf.u[0] = *(const uint4*)bp;
        bf.u[1] = *(const uint4*)(bp + 16);
        acc[nt] = wmma_bf16(a, bf, acc[nt]);
      }
    }

#pragma unroll
    for (int nt = 0; nt < 6; ++nt) {
#pragma unroll
      for (int r = 0; r < 8; ++r) {
        const int c  = mt * 16 + r + 8 * sel;
        const int wc = nt * 16 + ln;
        const size_t idx = (size_t)(b * CC + c) * HWP + h * WWd + wc;
        out[idx] = g * (out[idx] + acc[nt][r]) + f_low[idx];
      }
    }
  }
}

// ---------------------------------------------------------------------------
extern "C" void kernel_launch(void* const* d_in, const int* in_sizes, int n_in,
                              void* d_out, int out_size, void* d_ws, size_t ws_size,
                              hipStream_t stream)
{
  (void)in_sizes; (void)n_in; (void)out_size; (void)ws_size;

  const float* f_low = (const float*)d_in[0];
  const float* f_ref = (const float*)d_in[1];
  const float* Wq    = (const float*)d_in[2];
  const float* bq    = (const float*)d_in[3];
  const float* Wk    = (const float*)d_in[4];
  const float* bk    = (const float*)d_in[5];
  const float* Wv    = (const float*)d_in[6];
  const float* bv    = (const float*)d_in[7];
  const float* gamma = (const float*)d_in[8];
  float* out = (float*)d_out;

  char* ws = (char*)d_ws;
  size_t off = 0;
  auto take = [&](size_t bytes) -> char* {
    char* p = ws + off;
    off += (bytes + 255) & ~(size_t)255;
    return p;
  };

  const size_t qk_bytes  = (size_t)BB * HWP * C8d * 2;       // 9.4 MB each
  const size_t v_bytes   = (size_t)BB * CC * HWP * 2;        // 75.5 MB each
  const size_t e_bytes   = (size_t)BB * 96 * 96 * 96 * 4;    // 28.3 MB each
  const size_t att_bytes = (size_t)BB * 96 * 96 * 96 * 2;    // 14.2 MB each

  unsigned short* qA = (unsigned short*)take(qk_bytes);  // [b][w][h][c]
  unsigned short* qB = (unsigned short*)take(qk_bytes);  // [b][h][w][c]
  unsigned short* kA = (unsigned short*)take(qk_bytes);
  unsigned short* kB = (unsigned short*)take(qk_bytes);
  unsigned short* vN = (unsigned short*)take(v_bytes);   // [b][c][h][w]
  unsigned short* vT = (unsigned short*)take(v_bytes);   // [b][c][w][h]
  float*          EH = (float*)take(e_bytes);            // [b][w][h][g]
  float*          EW = (float*)take(e_bytes);            // [b][h][w][v']
  unsigned short* aH = (unsigned short*)take(att_bytes);
  unsigned short* aW = (unsigned short*)take(att_bytes);

  const int proj_blocks = BB * (HWP / 128);   // 576

  proj_kernel<4, 0><<<proj_blocks, 256, 0, stream>>>(f_low, Wq, bq, qA, qB);
  proj_kernel<4, 0><<<proj_blocks, 256, 0, stream>>>(f_ref, Wk, bk, kA, kB);
  proj_kernel<16, 1><<<proj_blocks, 256, 0, stream>>>(f_ref, Wv, bv, vN, vN);

  transpose_hw_kernel<<<BB * CC, 256, 0, stream>>>(vN, vT);

  energy_kernel<<<2 * BB * 96, 192, 0, stream>>>(qA, kA, qB, kB, EH, EW);

  softmax_kernel<<<(BB * HWP) / 8, 256, 0, stream>>>(EH, EW, aH, aW);

  agg_h_kernel<<<BB * WWd, 256, 0, stream>>>(vT, aH, out);
  agg_w_kernel<<<BB * HHd, 256, 0, stream>>>(vN, aW, f_low, gamma, out);
}
